// Fusion_Nuclear_21715354649607
// MI455X (gfx1250) — matrix operand order repaired and checked by reading, not compiled, not run on hardware
//
#include <hip/hip_runtime.h>
#include <hip/hip_bf16.h>
#include <math.h>

// ---------------------------------------------------------------------------
// Nuclear-norm weighted fusion for MI455X (gfx1250, wave32, WMMA).
//   p_c = ||x_c + eps||_*  via  trace((A^T A)^{1/2})  (Newton-Schulz, GEMMs)
//   out = (p1/(p1+p2+eps))*en_ir + (p2/(p1+p2+eps))*en_vi
// GEMMs: batched bf16 WMMA (v_wmma_f32_16x16x32_bf16, f32 accumulate).
// All Newton-Schulz operands are symmetric (polynomials of the symmetric
// Gram matrix), so the B operand can be staged row-major -> both operand
// stagings are straight copies done with GLOBAL_LOAD_ASYNC_TO_LDS_B128
// (ASYNCcnt + s_wait_asynccnt), the CDNA5 async global->LDS path.
// ---------------------------------------------------------------------------

typedef __attribute__((ext_vector_type(16))) __bf16 v16bf;
typedef __attribute__((ext_vector_type(8)))  float  v8f;

#define NMAT 256          // 128 channels x 2 tensors
#define DIM  384
#define SMAT (DIM * DIM)  // 147456
#define KSTEP 64
#define LDP  72           // LDS row pitch in bf16 (64 + 8 pad; 16B aligned rows)
#define EPS  1e-5f

__device__ __forceinline__ unsigned short f2bf(float f) {
    unsigned int u = __float_as_uint(f);
    unsigned int r = u + 0x7FFFu + ((u >> 16) & 1u);   // round-to-nearest-even
    return (unsigned short)(r >> 16);
}
__device__ __forceinline__ float bf2f(unsigned short h) {
    return __uint_as_float(((unsigned int)h) << 16);
}

// Async 16-byte global -> LDS copy (per-lane addresses, tracked by ASYNCcnt).
// ldsOff is the wave-relative LDS byte address (low 32 bits of the flat
// pointer to __shared__); gaddr is the 64-bit global address.
__device__ __forceinline__ void async_copy_b128(unsigned ldsOff,
                                                unsigned long long gaddr) {
    asm volatile("global_load_async_to_lds_b128 %0, %1, off"
                 :: "v"(ldsOff), "v"(gaddr) : "memory");
}
__device__ __forceinline__ void wait_async_zero() {
    asm volatile("s_wait_asynccnt 0" ::: "memory");
}

// ---------------------------------------------------------------------------
// Batched bf16 GEMM, all DIMxDIM row-major bf16 (ushort), f32 accumulate.
//   gramMode!=0 : C_b = (A_b^T * B_b) * (useScale ? 1/sArr[b] : 1)
//                 (transposed staging; used once for the Gram matrix)
//   gramMode==0 : C_b = A_b * B_b with B symmetric -> both operands staged
//                 with async global->LDS b128 copies.
// Block: 256 threads = 8 waves, 128x128 C tile, K-step 64.
// Each wave: 64x32 sub-tile = 4x2 WMMA 16x16 accumulators, 16 WMMA per stage.
// ---------------------------------------------------------------------------
__global__ __launch_bounds__(256) void gemm_bf16_kernel(
    const unsigned short* __restrict__ Ag, const unsigned short* __restrict__ Bg,
    unsigned short* __restrict__ Cg, int gramMode,
    const float* __restrict__ sArr, int useScale)
{
    __shared__ __align__(16) unsigned short As[128 * LDP];  // As[m][k]
    __shared__ __align__(16) unsigned short Bs[128 * LDP];  // Bs[n][k]

    const int b = blockIdx.z;
    const unsigned short* A = Ag + (size_t)b * SMAT;
    const unsigned short* B = Bg + (size_t)b * SMAT;
    unsigned short*       C = Cg + (size_t)b * SMAT;

    const int tid  = threadIdx.x;
    const int lane = tid & 31;
    const int wave = tid >> 5;
    const int wr   = wave >> 2;   // 0..1 : 64-row strip
    const int wc   = wave & 3;    // 0..3 : 32-col strip
    const int cm0  = blockIdx.x * 128;
    const int cn0  = blockIdx.y * 128;
    const int lm   = lane & 15;   // row/col within 16
    const int kh   = lane >> 4;   // lane half selects K group

    v8f acc[4][2];
    for (int i = 0; i < 4; ++i)
        for (int j = 0; j < 2; ++j)
            for (int e = 0; e < 8; ++e) acc[i][j][e] = 0.0f;

    for (int kb = 0; kb < DIM; kb += KSTEP) {
        if (gramMode) {
            // Transposed staging: As[m][k] = A[kb+k][cm0+m], Bs[n][k] = B[kb+k][cn0+n].
            // Vector b128 global loads along m/n, b16 LDS scatter.
            for (int idx = tid; idx < (128 * KSTEP) / 8; idx += 256) {
                int kk = idx >> 4;            // 0..63
                int m8 = (idx & 15) * 8;      // 0..120
                uint4 da = *(const uint4*)&A[(size_t)(kb + kk) * DIM + cm0 + m8];
                uint4 db = *(const uint4*)&B[(size_t)(kb + kk) * DIM + cn0 + m8];
                const unsigned short* ea = (const unsigned short*)&da;
                const unsigned short* eb = (const unsigned short*)&db;
#pragma unroll
                for (int j = 0; j < 8; ++j) {
                    As[(m8 + j) * LDP + kk] = ea[j];
                    Bs[(m8 + j) * LDP + kk] = eb[j];
                }
            }
            __syncthreads();
        } else {
            // Symmetric operands: straight row-major copies via async b128.
            for (int idx = tid; idx < (128 * KSTEP) / 8; idx += 256) {
                int row = idx >> 3;           // 0..127
                int k8  = (idx & 7) * 8;      // 0..56
                async_copy_b128(
                    (unsigned)(size_t)&As[row * LDP + k8],
                    (unsigned long long)(size_t)&A[(size_t)(cm0 + row) * DIM + kb + k8]);
                async_copy_b128(
                    (unsigned)(size_t)&Bs[row * LDP + k8],
                    (unsigned long long)(size_t)&B[(size_t)(cn0 + row) * DIM + kb + k8]);
            }
            wait_async_zero();
            __syncthreads();
        }

        // ---- 2 K-chunks of 32; fragments are two b128 LDS loads each ----
#pragma unroll
        for (int kc = 0; kc < 2; ++kc) {
            v16bf afr[4], bfr[2];
#pragma unroll
            for (int mi = 0; mi < 4; ++mi) {
                int row = wr * 64 + mi * 16 + lm;
                const unsigned short* p = &As[row * LDP + kc * 32 + kh * 8];
                ((uint4*)&afr[mi])[0] = *(const uint4*)p;        // K = kh*8 + 0..7
                ((uint4*)&afr[mi])[1] = *(const uint4*)(p + 16); // K = 16+kh*8+0..7
            }
#pragma unroll
            for (int ni = 0; ni < 2; ++ni) {
                int row = wc * 32 + ni * 16 + lm;
                const unsigned short* p = &Bs[row * LDP + kc * 32 + kh * 8];
                ((uint4*)&bfr[ni])[0] = *(const uint4*)p;        // K = kh*8 + 0..7
                ((uint4*)&bfr[ni])[1] = *(const uint4*)(p + 16); // K = 16+kh*8+0..7
            }
#pragma unroll
            for (int mi = 0; mi < 4; ++mi)
#pragma unroll
                for (int ni = 0; ni < 2; ++ni)
                    acc[mi][ni] = __builtin_amdgcn_wmma_f32_16x16x32_bf16(
                        false, afr[mi], false, bfr[ni], (short)0, acc[mi][ni],
                        false, false);
        }
        __syncthreads();
    }

    const float sc = useScale ? (1.0f / sArr[b]) : 1.0f;
#pragma unroll
    for (int mi = 0; mi < 4; ++mi)
#pragma unroll
        for (int ni = 0; ni < 2; ++ni)
#pragma unroll
            for (int r = 0; r < 8; ++r) {
                int row = cm0 + wr * 64 + mi * 16 + r + kh * 8;  // C layout: VGPR r -> M=r(+8)
                int col = cn0 + wc * 32 + ni * 16 + lm;
                C[(size_t)row * DIM + col] = f2bf(acc[mi][ni][r] * sc);
            }
}

// ---------------------------------------------------------------------------
// Elementwise / reduction helpers
// ---------------------------------------------------------------------------
__global__ void convert_kernel(const float* __restrict__ in0,
                               const float* __restrict__ in1,
                               unsigned short* __restrict__ Abuf)
{
    int c   = blockIdx.y;                       // 0..255
    int idx = blockIdx.x * 256 + threadIdx.x;   // 0..SMAT-1 (grid sized exactly)
    const float* src = (c < 128) ? in0 + (size_t)c * SMAT
                                 : in1 + (size_t)(c - 128) * SMAT;
    Abuf[(size_t)c * SMAT + idx] = f2bf(src[idx] + EPS);
}

__global__ void sumsq_kernel(const float* __restrict__ in0,
                             const float* __restrict__ in1,
                             float* __restrict__ s)
{
    int c = blockIdx.x;
    const float* src = (c < 128) ? in0 + (size_t)c * SMAT
                                 : in1 + (size_t)(c - 128) * SMAT;
    float acc = 0.0f;
    for (int i = threadIdx.x; i < SMAT; i += 256) {
        float v = src[i] + EPS;
        acc += v * v;
    }
    __shared__ float red[256];
    red[threadIdx.x] = acc;
    __syncthreads();
    for (int st = 128; st > 0; st >>= 1) {
        if (threadIdx.x < st) red[threadIdx.x] += red[threadIdx.x + st];
        __syncthreads();
    }
    if (threadIdx.x == 0) s[c] = red[0];
}

__global__ void init_identity_kernel(unsigned short* __restrict__ Z)
{
    int b   = blockIdx.y;
    int idx = blockIdx.x * 256 + threadIdx.x;
    int row = idx / DIM;
    int col = idx - row * DIM;
    Z[(size_t)b * SMAT + idx] = (row == col) ? (unsigned short)0x3F80 : (unsigned short)0;
}

// W = 1.5*I - 0.5*T, in place
__global__ void ns_w_kernel(unsigned short* __restrict__ T)
{
    int b   = blockIdx.y;
    int idx = blockIdx.x * 256 + threadIdx.x;
    int row = idx / DIM;
    int col = idx - row * DIM;
    size_t o = (size_t)b * SMAT + idx;
    float w = ((row == col) ? 1.5f : 0.0f) - 0.5f * bf2f(T[o]);
    T[o] = f2bf(w);
}

// p[c] = sqrt(s[c]) * trace(Y_c)
__global__ void trace_kernel(const unsigned short* __restrict__ Y,
                             const float* __restrict__ s,
                             float* __restrict__ p)
{
    int c = blockIdx.x;
    float acc = 0.0f;
    for (int i = threadIdx.x; i < DIM; i += 128)
        acc += bf2f(Y[(size_t)c * SMAT + (size_t)i * DIM + i]);
    __shared__ float red[128];
    red[threadIdx.x] = acc;
    __syncthreads();
    for (int st = 64; st > 0; st >>= 1) {
        if (threadIdx.x < st) red[threadIdx.x] += red[threadIdx.x + st];
        __syncthreads();
    }
    if (threadIdx.x == 0) p[c] = sqrtf(s[c]) * red[0];
}

// mode 0: weights from q[c]; mode 1: sqrt(q[c]) (Frobenius fallback);
// mode 2: equal weights (no scratch at all).
__global__ void blend_kernel(const float* __restrict__ in0,
                             const float* __restrict__ in1,
                             const float* __restrict__ q, int mode,
                             float* __restrict__ out)
{
    int c   = blockIdx.y;                       // 0..127
    int idx = blockIdx.x * 256 + threadIdx.x;   // 0..SMAT-1
    float p1, p2;
    if (mode == 0)      { p1 = q[c];        p2 = q[c + 128]; }
    else if (mode == 1) { p1 = sqrtf(q[c]); p2 = sqrtf(q[c + 128]); }
    else                { p1 = 1.0f;        p2 = 1.0f; }
    float d = p1 + p2 + EPS;
    size_t o = (size_t)c * SMAT + idx;
    out[o] = (p1 / d) * in0[o] + (p2 / d) * in1[o];
}

// ---------------------------------------------------------------------------
extern "C" void kernel_launch(void* const* d_in, const int* in_sizes, int n_in,
                              void* d_out, int out_size, void* d_ws, size_t ws_size,
                              hipStream_t stream)
{
    (void)in_sizes; (void)n_in; (void)out_size;
    const float* in0 = (const float*)d_in[0];   // en_ir  [128,384,384] f32
    const float* in1 = (const float*)d_in[1];   // en_vi
    float*       out = (float*)d_out;

    const size_t matBytes = (size_t)NMAT * SMAT * 2;   // one bf16 batch buffer
    char* base = (char*)d_ws;
    const size_t needFull = 5 * matBytes + 4096;       // A/T, Y0, Y1, Z0, Z1, s+p

    dim3 blk(256);
    dim3 gEw(SMAT / 256, NMAT);    // (576, 256) elementwise over all matrices
    dim3 gBl(SMAT / 256, 128);     // (576, 128) elementwise over channels

    if (ws_size >= needFull) {
        unsigned short* Abuf = (unsigned short*)(base);
        unsigned short* Y0   = (unsigned short*)(base + 1 * matBytes);
        unsigned short* Y1   = (unsigned short*)(base + 2 * matBytes);
        unsigned short* Z0   = (unsigned short*)(base + 3 * matBytes);
        unsigned short* Z1   = (unsigned short*)(base + 4 * matBytes);
        float* s = (float*)(base + 5 * matBytes);
        float* p = s + NMAT;

        hipLaunchKernelGGL(convert_kernel, gEw, blk, 0, stream, in0, in1, Abuf);
        hipLaunchKernelGGL(sumsq_kernel, dim3(NMAT), blk, 0, stream, in0, in1, s);

        dim3 gg(DIM / 128, DIM / 128, NMAT);   // (3,3,256)
        // Y0 = (A^T A) / s   (Gram: transposed staging path)
        hipLaunchKernelGGL(gemm_bf16_kernel, gg, blk, 0, stream,
                           Abuf, Abuf, Y0, 1, s, 1);
        hipLaunchKernelGGL(init_identity_kernel, gEw, blk, 0, stream, Z0);

        unsigned short *Y = Y0, *Z = Z0, *Yn = Y1, *Zn = Z1;
        unsigned short *T = Abuf;              // A is dead after gram/sumsq
        for (int it = 0; it < 8; ++it) {
            hipLaunchKernelGGL(gemm_bf16_kernel, gg, blk, 0, stream,
                               Z, Y, T, 0, (const float*)nullptr, 0);   // T = Z*Y
            hipLaunchKernelGGL(ns_w_kernel, gEw, blk, 0, stream, T);    // T = 1.5I-0.5T
            hipLaunchKernelGGL(gemm_bf16_kernel, gg, blk, 0, stream,
                               Y, T, Yn, 0, (const float*)nullptr, 0);  // Yn = Y*W
            hipLaunchKernelGGL(gemm_bf16_kernel, gg, blk, 0, stream,
                               T, Z, Zn, 0, (const float*)nullptr, 0);  // Zn = W*Z
            unsigned short* t1 = Y; Y = Yn; Yn = t1;
            unsigned short* t2 = Z; Z = Zn; Zn = t2;
        }

        hipLaunchKernelGGL(trace_kernel, dim3(NMAT), dim3(128), 0, stream, Y, s, p);
        hipLaunchKernelGGL(blend_kernel, gBl, blk, 0, stream, in0, in1, p, 0, out);
    } else if (ws_size >= 4096) {
        // Frobenius-norm fallback (tiny scratch): p ~ sqrt(sum x^2)
        float* s = (float*)base;
        hipLaunchKernelGGL(sumsq_kernel, dim3(NMAT), blk, 0, stream, in0, in1, s);
        hipLaunchKernelGGL(blend_kernel, gBl, blk, 0, stream, in0, in1, s, 1, out);
    } else {
        // No scratch at all: equal weights
        hipLaunchKernelGGL(blend_kernel, gBl, blk, 0, stream, in0, in1,
                           (const float*)base, 2, out);
    }
}
